// MSAttention_29059748725180
// MI455X (gfx1250) — matrix operand order repaired and checked by reading, not compiled, not run on hardware
//
#include <hip/hip_runtime.h>
#include <hip/hip_bf16.h>

// CDNA5 / gfx1250: wave32, WMMA 16x16x32 bf16.
typedef __attribute__((ext_vector_type(16))) __bf16 v16bf;
typedef __attribute__((ext_vector_type(8)))  float  v8f;

__device__ __forceinline__ unsigned short f32_to_bf16_rne(float f) {
    unsigned u = __builtin_bit_cast(unsigned, f);
    unsigned r = u + 0x7FFFu + ((u >> 16) & 1u);   // round-to-nearest-even
    return (unsigned short)(r >> 16);
}

struct WPtrs { const float* p[7]; };

// Convert the V-half (rows 256..511) of each [512,256] kv weight to bf16 [256,256].
__global__ __launch_bounds__(256) void prep_weights(WPtrs w, unsigned short* __restrict__ dst) {
    int id = blockIdx.x * 256 + threadIdx.x;          // 7 * 65536 total
    if (id >= 7 * 65536) return;
    int wi = id >> 16, r = id & 65535;
    dst[id] = f32_to_bf16_rne(w.p[wi][65536 + r]);    // row 256 starts at 256*256
}

// dst[b,y1,x1,c] (bf16) = sum over hb x hb block of src[b,H,W,256] (f32)
__global__ __launch_bounds__(256)
void blocksum_bf16(const float* __restrict__ src, unsigned short* __restrict__ dst,
                   int h1, int w1, int hb, int H, int W, int nOut) {
    int id = blockIdx.x * 256 + threadIdx.x;
    if (id >= nOut) return;
    int c = id & 255;
    int blk = id >> 8;
    int x1 = blk % w1, t = blk / w1;
    int y1 = t % h1,  b = t / h1;
    const float* base = src + (((size_t)b * H + (size_t)y1 * hb) * W + (size_t)x1 * hb) * 256 + c;
    float s = 0.f;
    for (int dy = 0; dy < hb; ++dy)
        for (int dx = 0; dx < hb; ++dx)
            s += base[((size_t)dy * W + dx) * 256];
    dst[id] = f32_to_bf16_rne(s);
}

// One wave32 per 16x16 output tile: C[rows,256] = A[rows,256] @ W1^T (+ Apar @ W2^T).
// A is bf16 (pre-reduced activations); Apar is raw f32 (parent pixels), converted inline.
// Results are written to outBase: hb==1 -> per-pixel row; hb>1 -> broadcast over hb x hb
// block (accumulate!=0 adds on top of previously written kid term).
__global__ __launch_bounds__(256)
void gemm_wmma(const unsigned short* __restrict__ A,
               const unsigned short* __restrict__ W1,
               const float* __restrict__ Apar,
               const unsigned short* __restrict__ W2,
               float* __restrict__ outBase,
               int rows, int h1, int w1, int hb, int H, int W, int accumulate) {
    const int lane   = threadIdx.x & 31;
    const int tile   = blockIdx.x * 8 + (threadIdx.x >> 5);  // 8 waves / block
    const int rowTile = tile >> 4;                           // 16 col-tiles (256 outs)
    const int colTile = tile & 15;
    if (rowTile * 16 >= rows) return;                        // whole-wave uniform

    const int m  = lane & 15;        // A row within tile / C column lane
    const int hi = lane >> 4;        // lane half (ISA striping)
    const int row  = rowTile * 16 + m;
    const int ocol = colTile * 16 + m;                       // B column = lane&15

    const unsigned short* aRow = A  + (size_t)row  * 256;
    const unsigned short* wRow = W1 + (size_t)ocol * 256;
    __builtin_prefetch(aRow, 0, 1);                          // global_prefetch_b8
    __builtin_prefetch(wRow, 0, 1);

    v8f acc = {};
    for (int kc = 0; kc < 256; kc += 32) {
        union { unsigned u[8]; v16bf v; } a, b;
        #pragma unroll
        for (int j = 0; j < 8; ++j) {
            // ISA 7.12.2: 16-bit A 16x32 — VGPR j holds K pair {ka,ka+1}
            int ka = (j < 4) ? (8 * hi + 2 * j) : (16 + 8 * hi + 2 * (j - 4));
            a.u[j] = *(const unsigned*)(aRow + kc + ka);
            // 16-bit B 32x16 — VGPR j holds rows {kb,kb+1} (same K order as A), col = lane&15
            int kb = (j < 4) ? (8 * hi + 2 * j) : (16 + 8 * hi + 2 * (j - 4));
            b.u[j] = *(const unsigned*)(wRow + kc + kb);
        }
        acc = __builtin_amdgcn_wmma_f32_16x16x32_bf16(false, a.v, false, b.v,
                                                      (short)0, acc, false, false);
    }
    if (Apar) {  // fused parent term: extends K to 512 on the same accumulator
        const float* pRow = Apar + (size_t)row * 256;
        const unsigned short* w2Row = W2 + (size_t)ocol * 256;
        __builtin_prefetch(pRow, 0, 1);
        for (int kc = 0; kc < 256; kc += 32) {
            union { unsigned u[8]; v16bf v; } a, b;
            #pragma unroll
            for (int j = 0; j < 8; ++j) {
                int ka = (j < 4) ? (8 * hi + 2 * j) : (16 + 8 * hi + 2 * (j - 4));
                unsigned lo = f32_to_bf16_rne(pRow[kc + ka]);
                unsigned hi16 = f32_to_bf16_rne(pRow[kc + ka + 1]);
                a.u[j] = lo | (hi16 << 16);
                int kb = (j < 4) ? (8 * hi + 2 * j) : (16 + 8 * hi + 2 * (j - 4));
                b.u[j] = *(const unsigned*)(w2Row + kc + kb);
            }
            acc = __builtin_amdgcn_wmma_f32_16x16x32_bf16(false, a.v, false, b.v,
                                                          (short)0, acc, false, false);
        }
    }
    // C layout: VGPR v, lanes 0-15 -> M=v, lanes 16-31 -> M=8+v; N = lane&15
    #pragma unroll
    for (int v = 0; v < 8; ++v) {
        int M = rowTile * 16 + v + 8 * hi;      // block (or pixel) index
        float cv = acc[v];
        if (hb == 1) {
            size_t o = (size_t)M * 256 + ocol;
            outBase[o] = accumulate ? (outBase[o] + cv) : cv;
        } else {
            int x1 = M % w1, t = M / w1;
            int y1 = t % h1, b = t / h1;
            for (int dy = 0; dy < hb; ++dy)
                for (int dx = 0; dx < hb; ++dx) {
                    size_t pix = ((size_t)b * H + (y1 * hb + dy)) * W + (x1 * hb + dx);
                    size_t o = pix * 256 + ocol;
                    outBase[o] = accumulate ? (outBase[o] + cv) : cv;
                }
        }
    }
}

extern "C" void kernel_launch(void* const* d_in, const int* in_sizes, int n_in,
                              void* d_out, int out_size, void* d_ws, size_t ws_size,
                              hipStream_t stream) {
    const float* x0 = (const float*)d_in[0];   // [8,16,16,256]
    const float* x1 = (const float*)d_in[1];   // [8,32,32,256]
    const float* x2 = (const float*)d_in[2];   // [8,64,64,256]
    // setup_inputs order: x0,x1,x2, (qpeer,kvpeer)x3, (qpar,kvpar)x{1,2}, (qkid,kvkid)x{0,1}
    const float* kvpeer0 = (const float*)d_in[4];
    const float* kvpeer1 = (const float*)d_in[6];
    const float* kvpeer2 = (const float*)d_in[8];
    const float* kvpar1  = (const float*)d_in[10];
    const float* kvpar2  = (const float*)d_in[12];
    const float* kvkid0  = (const float*)d_in[14];
    const float* kvkid1  = (const float*)d_in[16];

    // workspace: bf16 reduced activations + bf16 V-weights (~6 MB)
    unsigned short* A0blk = (unsigned short*)d_ws;     //  128*256 (4x4 blocksum of x0)
    unsigned short* A1blk = A0blk + 128  * 256;        // 2048*256 (2x2 blocksum of x1)
    unsigned short* A2blk = A1blk + 2048 * 256;        // 8192*256 (2x2 blocksum of x2)
    unsigned short* Wv    = A2blk + 8192 * 256;        // 7 x [256,256]
    unsigned short* Wpeer0 = Wv + 0 * 65536;
    unsigned short* Wpeer1 = Wv + 1 * 65536;
    unsigned short* Wpeer2 = Wv + 2 * 65536;
    unsigned short* Wpar1  = Wv + 3 * 65536;
    unsigned short* Wpar2  = Wv + 4 * 65536;
    unsigned short* Wkid0  = Wv + 5 * 65536;
    unsigned short* Wkid1  = Wv + 6 * 65536;

    float* out0 = (float*)d_out;
    float* out1 = out0 + 8 * 16 * 16 * 256;
    float* out2 = out1 + 8 * 32 * 32 * 256;

    WPtrs wp = {{kvpeer0, kvpeer1, kvpeer2, kvpar1, kvpar2, kvkid0, kvkid1}};
    prep_weights<<<(7 * 65536) / 256, 256, 0, stream>>>(wp, Wv);

    blocksum_bf16<<<(128  * 256) / 256, 256, 0, stream>>>(x0, A0blk, 4, 4, 4, 16, 16, 128 * 256);
    blocksum_bf16<<<(2048 * 256) / 256, 256, 0, stream>>>(x1, A1blk, 16, 16, 2, 32, 32, 2048 * 256);
    blocksum_bf16<<<(8192 * 256) / 256, 256, 0, stream>>>(x2, A2blk, 32, 32, 2, 64, 64, 8192 * 256);

    // Kid terms (per-pixel rows), plain store into out:
    gemm_wmma<<<2048 / 8, 256, 0, stream>>>(A1blk, Wkid0, nullptr, nullptr, out0,
                                            2048, 0, 0, 1, 0, 0, 0);
    gemm_wmma<<<8192 / 8, 256, 0, stream>>>(A2blk, Wkid1, nullptr, nullptr, out1,
                                            8192, 0, 0, 1, 0, 0, 0);
    // Peer (+ fused parent) terms, broadcast over hb x hb block:
    gemm_wmma<<<128 / 8, 256, 0, stream>>>(A0blk, Wpeer0, nullptr, nullptr, out0,
                                           128, 4, 4, 4, 16, 16, /*accum=*/1);
    gemm_wmma<<<2048 / 8, 256, 0, stream>>>(A1blk, Wpeer1, x0, Wpar1, out1,
                                            2048, 16, 16, 2, 32, 32, /*accum=*/1);
    gemm_wmma<<<8192 / 8, 256, 0, stream>>>(A2blk, Wpeer2, x1, Wpar2, out2,
                                            8192, 32, 32, 2, 64, 64, /*accum=*/0);
}
